// AttnDecoderRNN_88570815578625
// MI455X (gfx1250) — matrix operand order repaired and checked by reading, not compiled, not run on hardware
//
#include <hip/hip_runtime.h>
#include <math.h>

// ---------------------------------------------------------------------------
// Problem constants
// ---------------------------------------------------------------------------
#define VV 50257
#define HH 1024
#define LL 10
#define BB 256

typedef unsigned short u16;
typedef __attribute__((ext_vector_type(16))) __bf16        v16bf;
typedef __attribute__((ext_vector_type(8)))  float         v8f;
typedef __attribute__((ext_vector_type(4)))  unsigned int  uint4v;

union FragU { uint4v u[2]; v16bf v; };

// fp32 -> bf16 round-to-nearest-even (pure integer, no __bf16 scalar math)
__device__ __forceinline__ u16 f2bf(float f) {
    union { float f; unsigned u; } x; x.f = f;
    unsigned r = x.u + 0x7FFFu + ((x.u >> 16) & 1u);
    return (u16)(r >> 16);
}

// pack two f32 bit-patterns into one u32 holding two RNE bf16 values
__device__ __forceinline__ unsigned pack2bf(unsigned a, unsigned b) {
    unsigned ra = a + 0x7FFFu + ((a >> 16) & 1u);
    unsigned rb = b + 0x7FFFu + ((b >> 16) & 1u);
    return (ra >> 16) | (rb & 0xFFFF0000u);
}

// ---------------------------------------------------------------------------
// Front: embedding gather + attention(L=10) + softmax + attn_applied.
// One block (256 threads) per batch row. Writes:
//   attn_weights -> d_out slot, xin_bf = [embedded | attn_applied] (bf16),
//   h_bf (bf16 copy of incoming hidden state)
// ---------------------------------------------------------------------------
__global__ __launch_bounds__(256) void attn_front(
    const int* __restrict__ tokens, const float* __restrict__ hidden,
    const float* __restrict__ enc, const float* __restrict__ emb,
    const float* __restrict__ attn_W, const float* __restrict__ attn_b,
    float* __restrict__ attnw_out, u16* __restrict__ xin_bf, u16* __restrict__ h_bf)
{
    const int b = blockIdx.x;
    const int t = threadIdx.x;
    __shared__ float s_red[256];
    __shared__ float s_logit[16];

    const int tok = tokens[b];
    const float* er = emb + (size_t)tok * HH;
    const float* hr = hidden + (size_t)b * HH;

    float e[4], h4[4];
#pragma unroll
    for (int i = 0; i < 4; ++i) {
        int j = t * 4 + i;
        e[i]  = er[j];
        h4[i] = hr[j];
        xin_bf[(size_t)b * (2 * HH) + j] = f2bf(e[i]);
        h_bf [(size_t)b * HH + j]        = f2bf(h4[i]);
    }

    // attention logits: dot(concat(embedded, h), attn_W[l]) + attn_b[l]
    for (int l = 0; l < LL; ++l) {
        const float* w = attn_W + (size_t)l * (2 * HH);
        float p = 0.f;
#pragma unroll
        for (int i = 0; i < 4; ++i) {
            int j = t * 4 + i;
            p += e[i] * w[j] + h4[i] * w[HH + j];
        }
        s_red[t] = p; __syncthreads();
        for (int s = 128; s > 0; s >>= 1) {
            if (t < s) s_red[t] += s_red[t + s];
            __syncthreads();
        }
        if (t == 0) s_logit[l] = s_red[0] + attn_b[l];
        __syncthreads();
    }

    // softmax over L=10 (computed redundantly by all threads; tiny)
    float mx = -INFINITY;
#pragma unroll
    for (int l = 0; l < LL; ++l) mx = fmaxf(mx, s_logit[l]);
    float w10[LL]; float sum = 0.f;
#pragma unroll
    for (int l = 0; l < LL; ++l) { w10[l] = expf(s_logit[l] - mx); sum += w10[l]; }
    const float inv = 1.f / sum;
    if (t < LL) attnw_out[(size_t)b * LL + t] = w10[t] * inv;

    // attn_applied = attn_weights @ encoder_outputs  -> second half of xin
#pragma unroll
    for (int i = 0; i < 4; ++i) {
        int j = t * 4 + i;
        float a = 0.f;
#pragma unroll
        for (int l = 0; l < LL; ++l) a += (w10[l] * inv) * enc[(size_t)l * HH + j];
        xin_bf[(size_t)b * (2 * HH) + HH + j] = f2bf(a);
    }
}

// ---------------------------------------------------------------------------
// WMMA A-fragment load (ISA 7.12.2, wave32). LD is a compile-time leading
// dimension so tile offsets fold into the 24-bit instruction offset field
// (one base VGPR pair for all MT tiles; no per-tile 64-bit VALU adds).
//   lanes 0-15  row M=lane,    K = k0..k0+15
//   lanes 16-31 row M=lane-16, K = k0+16..k0+31
// ---------------------------------------------------------------------------
template <int LD>
__device__ __forceinline__ v16bf load_a_frag(const u16* A, int row, int k0, int half) {
    const u16* p = A + (size_t)row * LD + k0 + half * 16;
    FragU f;
    f.u[0] = *(const uint4v*)(p);
    f.u[1] = *(const uint4v*)(p + 8);
    return f.v;
}

// ---------------------------------------------------------------------------
// B-fragment raw load: 16 contiguous f32 weights (row `col`, K=k0+half*16..)
// NON-TEMPORAL (streaming; keep L2 for logits/activations).
//   lanes 0-15  col N=lane,    K = k0..k0+15
//   lanes 16-31 col N=lane-16, K = k0+16..k0+31
// ---------------------------------------------------------------------------
template <int LD>
__device__ __forceinline__ void load_b_raw(const float* W, int col, int k0,
                                           int half, uint4v w[4]) {
    const uint4v* p = (const uint4v*)(W + (size_t)col * LD + k0 + half * 16);
    w[0] = __builtin_nontemporal_load(p + 0);
    w[1] = __builtin_nontemporal_load(p + 1);
    w[2] = __builtin_nontemporal_load(p + 2);
    w[3] = __builtin_nontemporal_load(p + 3);
}

__device__ __forceinline__ v16bf cvt_b(const uint4v w[4]) {
    FragU f;
    f.u[0][0] = pack2bf(w[0][0], w[0][1]);
    f.u[0][1] = pack2bf(w[0][2], w[0][3]);
    f.u[0][2] = pack2bf(w[1][0], w[1][1]);
    f.u[0][3] = pack2bf(w[1][2], w[1][3]);
    f.u[1][0] = pack2bf(w[2][0], w[2][1]);
    f.u[1][1] = pack2bf(w[2][2], w[2][3]);
    f.u[1][2] = pack2bf(w[3][0], w[3][1]);
    f.u[1][3] = pack2bf(w[3][2], w[3][3]);
    return f.v;
}

// ---------------------------------------------------------------------------
// GEMM: out[256, N] = A[256, K](bf16) @ W[N, K](f32, cvt on the fly)^T + bias
// One wave per block; wave owns MT accumulator tiles (MT*16 rows) x 16 cols.
// Software pipeline: B (HBM, NT) double-buffered one k-step ahead; A (L2)
// fragments prefetched one tile ahead. EXEC all-ones through every WMMA.
// grid = (ceil(N/16), 256/(MT*16)). MT=16 => weights cross HBM exactly once.
// ---------------------------------------------------------------------------
template <int MT, int LD>
__global__ __launch_bounds__(32) void wmma_gemm_xwT(
    const u16* __restrict__ A,
    const float* __restrict__ W,
    const float* __restrict__ bias,
    float* __restrict__ outF, u16* __restrict__ outB,
    int N, int ldo, int K, int relu)
{
    const int lane = threadIdx.x & 31;
    const int r    = lane & 15;
    const int half = lane >> 4;
    const int nbase = blockIdx.x * 16;
    const int mbase = blockIdx.y * (MT * 16);
    int col = nbase + r;
    if (col > N - 1) col = N - 1;          // clamp loads; stores masked later

    v8f acc[MT];
#pragma unroll
    for (int i = 0; i < MT; ++i)
#pragma unroll
        for (int j = 0; j < 8; ++j) acc[i][j] = 0.f;

    uint4v wcur[4], wnxt[4];
    load_b_raw<LD>(W, col, 0, half, wcur);
    v16bf a_cur = load_a_frag<LD>(A, mbase + r, 0, half);

    for (int k0 = 0; k0 < K; k0 += 32) {
        const int kn = (k0 + 32 < K) ? (k0 + 32) : 0;   // wrap: dummy reload of k=0
        load_b_raw<LD>(W, col, kn, half, wnxt);          // hide HBM latency
        const v16bf bf = cvt_b(wcur);
#pragma unroll
        for (int mt = 0; mt < MT; ++mt) {
            // prefetch next A fragment (next tile, or tile 0 of next k-step)
            v16bf a_nxt = (mt + 1 < MT)
                ? load_a_frag<LD>(A, mbase + (mt + 1) * 16 + r, k0, half)
                : load_a_frag<LD>(A, mbase + r, kn, half);
            acc[mt] = __builtin_amdgcn_wmma_f32_16x16x32_bf16(
                false, a_cur, false, bf, (short)0, acc[mt], false, false);
            a_cur = a_nxt;
        }
#pragma unroll
        for (int i = 0; i < 4; ++i) wcur[i] = wnxt[i];
    }

    const int n = nbase + r;
    if (n < N) {
        const float bv = bias ? bias[n] : 0.f;
#pragma unroll
        for (int mt = 0; mt < MT; ++mt) {
#pragma unroll
            for (int i = 0; i < 8; ++i) {
                const int m = mbase + mt * 16 + half * 8 + i;
                float v = acc[mt][i] + bv;
                if (relu) v = fmaxf(v, 0.f);
                if (outF) outF[(size_t)m * ldo + n] = v;
                if (outB) outB[(size_t)m * ldo + n] = f2bf(v);
            }
        }
    }
}

// ---------------------------------------------------------------------------
// GRU gate combine (PyTorch r,z,n ordering). Writes h_new (f32 -> d_out slot)
// and a bf16 copy for the logits GEMM.
// ---------------------------------------------------------------------------
__global__ __launch_bounds__(256) void gru_combine(
    const float* __restrict__ gi, const float* __restrict__ gh,
    const float* __restrict__ hidden,
    float* __restrict__ hnew_out, u16* __restrict__ hnew_bf)
{
    int idx = blockIdx.x * 256 + threadIdx.x;      // B*H threads
    if (idx >= BB * HH) return;
    const int b = idx >> 10;
    const int j = idx & (HH - 1);
    const float* gib = gi + (size_t)b * (3 * HH);
    const float* ghb = gh + (size_t)b * (3 * HH);
    const float r = 1.f / (1.f + expf(-(gib[j]            + ghb[j])));
    const float z = 1.f / (1.f + expf(-(gib[HH + j]       + ghb[HH + j])));
    const float n = tanhf(gib[2 * HH + j] + r * ghb[2 * HH + j]);
    const float hv = hidden[idx];
    const float hn = (1.f - z) * n + z * hv;
    hnew_out[idx] = hn;
    hnew_bf[idx]  = f2bf(hn);
}

// ---------------------------------------------------------------------------
// In-place row-wise log_softmax over N=V. One block per row.
// Pass 1: single sweep online (max, sum-exp). Pass 2: subtract lse.
// Logits should be L2-resident (weight stream was non-temporal).
// ---------------------------------------------------------------------------
__global__ __launch_bounds__(256) void log_softmax_rows(float* __restrict__ logits, int N)
{
    const int b = blockIdx.x;
    const int t = threadIdx.x;
    float* row = logits + (size_t)b * N;
    __shared__ float smx[256];
    __shared__ float ssum[256];

    float mx = -INFINITY, sum = 0.f;
    for (int j = t; j < N; j += 256) {
        float v = row[j];
        if (v > mx) { sum *= expf(mx - v); mx = v; }
        sum += expf(v - mx);
    }
    smx[t] = mx; ssum[t] = sum; __syncthreads();
    for (int s = 128; s > 0; s >>= 1) {
        if (t < s) {
            float m1 = smx[t], m2 = smx[t + s];
            float M  = fmaxf(m1, m2);
            ssum[t]  = ssum[t] * expf(m1 - M) + ssum[t + s] * expf(m2 - M);
            smx[t]   = M;
        }
        __syncthreads();
    }
    const float lse = smx[0] + logf(ssum[0]);
    __syncthreads();

    for (int j = t; j < N; j += 256) row[j] -= lse;
}

// ---------------------------------------------------------------------------
// Host launcher
// ---------------------------------------------------------------------------
extern "C" void kernel_launch(void* const* d_in, const int* in_sizes, int n_in,
                              void* d_out, int out_size, void* d_ws, size_t ws_size,
                              hipStream_t stream) {
    (void)in_sizes; (void)n_in; (void)out_size; (void)ws_size;

    const int*   tokens = (const int*)  d_in[0];
    const float* hidden = (const float*)d_in[1];
    const float* enc    = (const float*)d_in[2];
    const float* emb    = (const float*)d_in[3];
    const float* attn_W = (const float*)d_in[4];
    const float* attn_b = (const float*)d_in[5];
    const float* comb_W = (const float*)d_in[6];
    const float* comb_b = (const float*)d_in[7];
    const float* W_ih   = (const float*)d_in[8];
    const float* W_hh   = (const float*)d_in[9];
    const float* b_ih   = (const float*)d_in[10];
    const float* b_hh   = (const float*)d_in[11];
    const float* out_W  = (const float*)d_in[12];
    const float* out_b  = (const float*)d_in[13];

    float* out = (float*)d_out;
    float* out_logits = out;                                     // [B, V]
    float* out_hnew   = out + (size_t)BB * VV;                   // [1, B, H]
    float* out_attnw  = out + (size_t)BB * VV + (size_t)BB * HH; // [B, L]

    // workspace carve-out (256B aligned chunks) — ~9 MB total
    char* ws = (char*)d_ws;
    size_t off = 0;
    auto carve = [&](size_t bytes) -> void* {
        void* p = ws + off;
        off += (bytes + 255) & ~(size_t)255;
        return p;
    };
    u16*   xinb  = (u16*)  carve((size_t)BB * 2 * HH * 2);   // [B, 2H] bf16
    u16*   hb    = (u16*)  carve((size_t)BB * HH * 2);       // [B, H]  bf16
    u16*   xb    = (u16*)  carve((size_t)BB * HH * 2);       // [B, H]  bf16
    u16*   hnewb = (u16*)  carve((size_t)BB * HH * 2);       // [B, H]  bf16
    float* gi    = (float*)carve((size_t)BB * 3 * HH * 4);   // [B, 3H] f32
    float* gh    = (float*)carve((size_t)BB * 3 * HH * 4);   // [B, 3H] f32

    // 1) embedding + attention front (writes attn_weights slot of d_out)
    attn_front<<<BB, 256, 0, stream>>>(tokens, hidden, enc, emb, attn_W, attn_b,
                                       out_attnw, xinb, hb);

    // 2) x = relu(xin @ comb_W^T + comb_b)   [256,1024], K=2048  (bf16 out)
    wmma_gemm_xwT<4, 2 * HH><<<dim3(HH / 16, 4), 32, 0, stream>>>(
        xinb, comb_W, comb_b, nullptr, xb, HH, HH, 2 * HH, 1);

    // 3) gi = x @ W_ih^T + b_ih ; gh = h @ W_hh^T + b_hh   [256,3072], K=1024
    wmma_gemm_xwT<4, HH><<<dim3(3 * HH / 16, 4), 32, 0, stream>>>(
        xb, W_ih, b_ih, gi, nullptr, 3 * HH, 3 * HH, HH, 0);
    wmma_gemm_xwT<4, HH><<<dim3(3 * HH / 16, 4), 32, 0, stream>>>(
        hb, W_hh, b_hh, gh, nullptr, 3 * HH, 3 * HH, HH, 0);

    // 4) GRU combine -> h_new (f32 to d_out, bf16 to ws)
    gru_combine<<<(BB * HH) / 256, 256, 0, stream>>>(gi, gh, hidden, out_hnew, hnewb);

    // 5) logits = h_new @ out_W^T + out_b   [256, 50257], K=1024
    //    MT=16: out_W streamed from HBM exactly once (f32, NT), cvt in-register.
    wmma_gemm_xwT<16, HH><<<dim3((VV + 15) / 16, 1), 32, 0, stream>>>(
        hnewb, out_W, out_b, out_logits, nullptr, VV, VV, HH, 0);

    // 6) in-place log_softmax over V per row (online pass + subtract)
    log_softmax_rows<<<BB, 256, 0, stream>>>(out_logits, VV);
}
